// LSTMClassifierFX_73993696576078
// MI455X (gfx1250) — matrix operand order repaired and checked by reading, not compiled, not run on hardware
//
#include <hip/hip_runtime.h>
#include <hip/hip_bf16.h>

typedef __attribute__((ext_vector_type(16))) __bf16 v16bf;
typedef __attribute__((ext_vector_type(8)))  float  v8f;

#define HSZ   200
#define KPAD  224   // 7 K-tiles of 32
#define NGATE 4
#define SBLK  13    // 13 column blocks of 16 -> 208 padded columns per gate
#define NWAVE 13
#define NTHR  (NWAVE*32)
#define CHUNK 128

// ---------------------------------------------------------------------------
// Pack w_hh [800,200] f32 (row = gate output g, col = k) into bf16 WMMA
// B-fragment layout: tile t = q*13 + s covers gate q, columns j = s*16..s*16+15.
// Fragment (t, ktile): lane L holds column n = L&15 (j = s*16 + n),
// K = ktile*32 + (L>>4)*16 + e, e = 0..15 packed 2-per-dword.
// Flat layout: ((t*7 + ktile)*32 + lane) * 16 bf16  (1 KB per fragment).
// ---------------------------------------------------------------------------
__global__ void pack_b_kernel(const float* __restrict__ w_hh,
                              __bf16* __restrict__ Bpack) {
    const int frag = blockIdx.x;        // tile*7 + ktile, 52*7 = 364 frags
    const int tile = frag / 7;
    const int kt   = frag % 7;
    const int q    = tile / SBLK;
    const int s    = tile % SBLK;
    const int lane = threadIdx.x;       // 0..31 (one wave)
    const int nloc = lane & 15;
    const int half = lane >> 4;
    const int j    = s * 16 + nloc;     // column within gate (0..207)

    v16bf vals;
#pragma unroll
    for (int e = 0; e < 16; ++e) {
        const int k = kt * 32 + half * 16 + e;
        float v = 0.0f;
        if (j < HSZ && k < HSZ) v = w_hh[(q * HSZ + j) * HSZ + k];
        vals[e] = (__bf16)v;
    }
    uint4* dst = (uint4*)Bpack + ((size_t)frag * 32 + lane) * 2;
    dst[0] = ((uint4*)&vals)[0];
    dst[1] = ((uint4*)&vals)[1];
}

// ---------------------------------------------------------------------------
// Recurrent LSTM kernel: one workgroup per 16-batch tile, 13 waves.
// Wave s owns hidden columns [16s, 16s+16) and computes gates i,f,g,o there.
//   Gate 0,1 (i,f): B fragments register-resident for the whole kernel.
//   Gate 2   (g)  : B fragments staged once into LDS, read per step (DScnt).
//   Gate 3   (o)  : B fragments streamed from L2 per step (LOADcnt),
//                   1-deep software pipeline.
// This splits per-step weight traffic across the LDS and VMEM paths.
// Opaque zero offsets (empty asm) keep all streamed addresses loop-variant so
// LICM cannot hoist the loads (which would re-create register spills).
// ---------------------------------------------------------------------------
__global__ __launch_bounds__(NTHR, 1)
void lstm_kernel(const float* __restrict__ x,
                 const float* __restrict__ w_ih,
                 const float* __restrict__ b_ih,
                 const float* __restrict__ b_hh,
                 const float* __restrict__ fc_w,
                 const float* __restrict__ fc_b,
                 const __bf16* __restrict__ Bpack,
                 float* __restrict__ out,
                 int T) {
    __shared__ __bf16 hbuf[2][16][KPAD];        // h_t, bf16, double buffered
    __shared__ float  xbuf[16][CHUNK];          // staged input chunk
    __shared__ float  fcpart[2][NWAVE][16][4];  // per-wave fc partial sums
    __shared__ uint4  bls[NWAVE * 7 * 32 * 2];  // gate-2 fragments, 93 KB

    const int tid   = threadIdx.x;
    const int wave  = tid >> 5;          // 0..12  == column block s
    const int lane  = tid & 31;
    const int nloc  = lane & 15;
    const int half  = lane >> 4;
    const int mbase = half * 8;          // rows owned by this lane (C/D layout)
    const int j     = wave * 16 + nloc;  // hidden column owned (0..207)
    const int bblk  = blockIdx.x * 16;
    const bool jok  = (j < HSZ);

    // ---- register-resident B fragments for gates 0,1 (14 x 8 VGPRs) ----
    v16bf Breg[2 * 7];
#pragma unroll
    for (int q = 0; q < 2; ++q) {
#pragma unroll
        for (int k = 0; k < 7; ++k) {
            const int tile = q * SBLK + wave;
            const uint4* p = (const uint4*)Bpack + ((size_t)(tile * 7 + k) * 32 + lane) * 2;
            v16bf f;
            ((uint4*)&f)[0] = p[0];
            ((uint4*)&f)[1] = p[1];
            Breg[q * 7 + k] = f;
        }
    }
    // ---- stage gate-2 fragments into LDS (2912 x 32B chunks) ----
    // global chunk base for gate 2 = (2*SBLK)*7*32 = 5824; LDS chunk i maps to
    // global chunk 5824+i (identical internal ordering: ((w*7+k)*32+lane)).
    {
        const uint4* src = (const uint4*)Bpack + (size_t)(2 * SBLK * 7 * 32) * 2;
        for (int i = tid; i < NWAVE * 7 * 32; i += NTHR) {
            bls[i * 2]     = src[(size_t)i * 2];
            bls[i * 2 + 1] = src[(size_t)i * 2 + 1];
        }
    }
    // ---- streamed fragment base pointer for gate 3 (global) ----
    const uint4* bg3 = (const uint4*)Bpack + ((size_t)(3 * SBLK + wave) * 7 * 32 + lane) * 2;
    // ---- LDS fragment base (gate 2): chunk (wave*7 + k)*32 + lane ----
    const uint4* lg2 = bls + ((size_t)(wave * 7) * 32 + lane) * 2;

    // ---- per-lane scalars: input weights, biases, fc weights ----
    float wih[4], bias[4], fcw[4];
#pragma unroll
    for (int q = 0; q < NGATE; ++q) {
        const int g = q * HSZ + j;
        wih[q]  = jok ? w_ih[g] : 0.0f;
        bias[q] = jok ? (b_ih[g] + b_hh[g]) : 0.0f;
    }
#pragma unroll
    for (int o = 0; o < 4; ++o) fcw[o] = jok ? fc_w[o * HSZ + j] : 0.0f;
    const float fb = fc_b[tid & 3];      // used by tid < 64 reducers

    // ---- zero h buffers (h0 = 0, and K padding must be clean) ----
    for (int i = tid; i < 2 * 16 * KPAD; i += NTHR) ((__bf16*)hbuf)[i] = (__bf16)0.0f;

    float c[8];
#pragma unroll
    for (int r = 0; r < 8; ++r) c[r] = 0.0f;

    __syncthreads();

    for (int t = 0; t < T; ++t) {
        const int tc = t & (CHUNK - 1);
        if (tc == 0) {   // stage next x chunk (coalesced along t)
            for (int i = tid; i < 16 * CHUNK; i += NTHR) {
                const int m  = i >> 7;
                const int ti = i & (CHUNK - 1);
                const int tt = t + ti;
                xbuf[m][ti] = (tt < T) ? x[(size_t)(bblk + m) * T + tt] : 0.0f;
            }
            __syncthreads();
        }

        // deferred fc output store for step t-1 (13-way fixed-order sum)
        if (t > 0 && tid < 64) {
            const int m = tid >> 2, o = tid & 3;
            float s = fb;
#pragma unroll
            for (int w = 0; w < NWAVE; ++w) s += fcpart[(t - 1) & 1][w][m][o];
            out[((size_t)(bblk + m) * T + (t - 1)) * 4 + o] = s;
        }

        const int cur = t & 1, nxt = cur ^ 1;

        // Opaque zero offset: keeps streamed-weight addresses loop-variant so
        // LICM cannot hoist the loads out of the t-loop (no spills), while
        // preserving address-space provenance (global_load / ds_load).
        unsigned zoff = 0;
        asm volatile("" : "+v"(zoff));
        const uint4* p3 = bg3 + zoff;   // gate 3: global (L2)
        const uint4* l2 = lg2 + zoff;   // gate 2: LDS

        // ---- gates = h_{t-1} @ w_hh^T  via 28 WMMAs ----
        v8f acc[NGATE] = {};
        v16bf b3;
        ((uint4*)&b3)[0] = p3[0];
        ((uint4*)&b3)[1] = p3[1];
#pragma unroll
        for (int k = 0; k < 7; ++k) {
            v16bf n3;
            if (k < 6) {   // 1-deep pipeline for the global-streamed gate
                ((uint4*)&n3)[0] = p3[(size_t)(k + 1) * 64];
                ((uint4*)&n3)[1] = p3[(size_t)(k + 1) * 64 + 1];
            }
            // gate-2 fragment from LDS (low latency)
            v16bf b2;
            ((uint4*)&b2)[0] = l2[(size_t)k * 64];
            ((uint4*)&b2)[1] = l2[(size_t)k * 64 + 1];

            // A fragment of h: lane holds row nloc, K = k*32 + half*16 + {0..15}
            const __bf16* hrow = &hbuf[cur][nloc][k * 32 + half * 16];
            v16bf a;
            ((uint4*)&a)[0] = *(const uint4*)(hrow);
            ((uint4*)&a)[1] = *(const uint4*)(hrow + 8);

            acc[0] = __builtin_amdgcn_wmma_f32_16x16x32_bf16(
                false, a, false, Breg[0 * 7 + k], (short)0, acc[0], false, false);
            acc[1] = __builtin_amdgcn_wmma_f32_16x16x32_bf16(
                false, a, false, Breg[1 * 7 + k], (short)0, acc[1], false, false);
            acc[2] = __builtin_amdgcn_wmma_f32_16x16x32_bf16(
                false, a, false, b2, (short)0, acc[2], false, false);
            acc[3] = __builtin_amdgcn_wmma_f32_16x16x32_bf16(
                false, a, false, b3, (short)0, acc[3], false, false);

            if (k < 6) b3 = n3;
        }

        // ---- elementwise LSTM update for 8 rows x 1 column ----
#pragma unroll
        for (int r = 0; r < 8; ++r) {
            const int   m  = mbase + r;
            const float xr = xbuf[m][tc];
            const float gi = acc[0][r] + xr * wih[0] + bias[0];
            const float gf = acc[1][r] + xr * wih[1] + bias[1];
            const float gg = acc[2][r] + xr * wih[2] + bias[2];
            const float go = acc[3][r] + xr * wih[3] + bias[3];
            const float iv = 1.0f / (1.0f + __expf(-gi));
            const float fv = 1.0f / (1.0f + __expf(-gf));
            const float eg = __expf(2.0f * gg);
            const float gv = 1.0f - 2.0f / (eg + 1.0f);   // tanh
            const float ov = 1.0f / (1.0f + __expf(-go));
            const float cn = fv * c[r] + iv * gv;
            c[r] = cn;
            const float ec = __expf(2.0f * cn);
            const float th = 1.0f - 2.0f / (ec + 1.0f);   // tanh(c)
            const float hv = ov * th;
            hbuf[nxt][m][j] = (__bf16)hv;

            // fc partials for this row: reduce across the wave's 16 columns
            float p0 = hv * fcw[0];
            float p1 = hv * fcw[1];
            float q2 = hv * fcw[2];
            float q3 = hv * fcw[3];
            p0 += __shfl_xor(p0, 1); p0 += __shfl_xor(p0, 2);
            p0 += __shfl_xor(p0, 4); p0 += __shfl_xor(p0, 8);
            p1 += __shfl_xor(p1, 1); p1 += __shfl_xor(p1, 2);
            p1 += __shfl_xor(p1, 4); p1 += __shfl_xor(p1, 8);
            q2 += __shfl_xor(q2, 1); q2 += __shfl_xor(q2, 2);
            q2 += __shfl_xor(q2, 4); q2 += __shfl_xor(q2, 8);
            q3 += __shfl_xor(q3, 1); q3 += __shfl_xor(q3, 2);
            q3 += __shfl_xor(q3, 4); q3 += __shfl_xor(q3, 8);
            if (nloc == 0) {
                fcpart[cur][wave][m][0] = p0;
                fcpart[cur][wave][m][1] = p1;
                fcpart[cur][wave][m][2] = q2;
                fcpart[cur][wave][m][3] = q3;
            }
        }

        __syncthreads();
    }

    // final fc store for t = T-1
    if (tid < 64) {
        const int m = tid >> 2, o = tid & 3;
        float s = fb;
#pragma unroll
        for (int w = 0; w < NWAVE; ++w) s += fcpart[(T - 1) & 1][w][m][o];
        out[((size_t)(bblk + m) * T + (T - 1)) * 4 + o] = s;
    }
}

// ---------------------------------------------------------------------------
extern "C" void kernel_launch(void* const* d_in, const int* in_sizes, int n_in,
                              void* d_out, int out_size, void* d_ws, size_t ws_size,
                              hipStream_t stream) {
    const float* x    = (const float*)d_in[0];
    const float* w_ih = (const float*)d_in[1];
    const float* w_hh = (const float*)d_in[2];
    const float* b_ih = (const float*)d_in[3];
    const float* b_hh = (const float*)d_in[4];
    const float* fc_w = (const float*)d_in[5];
    const float* fc_b = (const float*)d_in[6];
    float*       out  = (float*)d_out;
    __bf16*      Bpk  = (__bf16*)d_ws;   // 52*7*32*16 bf16 = 364 KB

    const int B = 128;
    const int T = in_sizes[0] / B;       // 5000

    pack_b_kernel<<<52 * 7, 32, 0, stream>>>(w_hh, Bpk);
    lstm_kernel<<<B / 16, NTHR, 0, stream>>>(x, w_ih, b_ih, b_hh, fc_w, fc_b,
                                             Bpk, out, T);
}
